// SensitiveEstimator_54099408060633
// MI455X (gfx1250) — matrix operand order repaired and checked by reading, not compiled, not run on hardware
//
#include <hip/hip_runtime.h>
#include <hip/hip_bf16.h>

#define DIN 128
#define DH  64
#define BPACK_ELEMS (4 * 4 * 32 * 16)   // tn(4) x kstep(4) x lane(32) x 16 bf16 = 8192

typedef __attribute__((ext_vector_type(16))) __bf16 v16bf;
typedef __attribute__((ext_vector_type(8)))  float  v8f;

__device__ __forceinline__ unsigned short f2bf(float f) {
  unsigned int u = __float_as_uint(f);
  unsigned int r = u + 0x7fffu + ((u >> 16) & 1u);   // round-to-nearest-even
  return (unsigned short)(r >> 16);
}
__device__ __forceinline__ float bf2f(unsigned short b) {
  return __uint_as_float(((unsigned int)b) << 16);
}
__device__ __forceinline__ unsigned int wang_hash(unsigned int s) {
  s = (s ^ 61u) ^ (s >> 16);
  s *= 9u;
  s ^= s >> 4;
  s *= 0x27d4eb2du;
  s ^= s >> 15;
  return s;
}

// ---------------- normalization ----------------
__global__ void k_init_deg(float* deg, int n) {
  int i = blockIdx.x * blockDim.x + threadIdx.x;
  if (i < n) deg[i] = 1.0f;                           // self-loop weight
}
__global__ void k_deg_scatter(const int* dst, const float* w, float* deg, int e) {
  int i = blockIdx.x * blockDim.x + threadIdx.x;
  if (i < e) atomicAdd(&deg[dst[i]], w[i]);
}
__global__ void k_dinv(const float* deg, float* dinv, int n) {
  int i = blockIdx.x * blockDim.x + threadIdx.x;
  if (i < n) {
    float d = deg[i];
    dinv[i] = d > 0.0f ? __frsqrt_rn(d) : 0.0f;
  }
}

// ---------------- prep: x -> bf16, W1 -> WMMA B-register image ----------------
__global__ void k_pack_x(const float* __restrict__ x, unsigned short* __restrict__ xb,
                         int total) {
  int i = blockIdx.x * blockDim.x + threadIdx.x;
  if (i < total) xb[i] = f2bf(x[i]);
}
// Bpack[((tn*4+kc)*32+lane)*16 + j] = bf16(W1[(kc*32 + (lane>>4)*16 + j)*DH + tn*16 + (lane&15)])
__global__ void k_pack_B(const float* __restrict__ W1, unsigned short* __restrict__ Bpack) {
  int i = blockIdx.x * blockDim.x + threadIdx.x;      // over BPACK_ELEMS
  if (i >= BPACK_ELEMS) return;
  int j    = i & 15;
  int lane = (i >> 4) & 31;
  int kc   = (i >> 9) & 3;
  int tn   = (i >> 11) & 3;
  int k    = kc * 32 + (lane >> 4) * 16 + j;
  int col  = tn * 16 + (lane & 15);
  Bpack[i] = f2bf(W1[k * DH + col]);
}

// ---------------- h = x @ W1 (bf16 WMMA, f32 acc, h stored bf16) ----------------
// One wave per 16x16 tile. B image staged block-wide in LDS (16 KB).
__global__ void k_gemm_wmma(const unsigned short* __restrict__ xb,
                            const unsigned short* __restrict__ Bpack,
                            unsigned short* __restrict__ hb, int n) {
  __shared__ unsigned short Bs[BPACK_ELEMS];          // 16 KB
  // cooperative copy: 1024 x int4 = 16 KB
  for (int t = threadIdx.x; t < BPACK_ELEMS / 8; t += blockDim.x)
    reinterpret_cast<int4*>(Bs)[t] = reinterpret_cast<const int4*>(Bpack)[t];
  __syncthreads();

  const int wave = (blockIdx.x * blockDim.x + threadIdx.x) >> 5;
  const int lane = threadIdx.x & 31;
  const int tm = wave >> 2;                           // row tile
  const int tn = wave & 3;                            // col tile
  if (tm >= n / 16) return;                           // wave-uniform (grid is exact anyway)

  const int l15   = lane & 15;
  const int khalf = lane >> 4;
  const int row   = tm * 16 + l15;
  const int col   = tn * 16 + l15;
  const unsigned short* xrow = xb + (size_t)row * DIN;
  __builtin_prefetch(xrow, 0, 0);                     // global_prefetch_b8

  v8f c = {};
  #pragma unroll
  for (int kc = 0; kc < 4; ++kc) {
    const int k0 = kc * 32;
    // A: two contiguous 8-bf16 (16 B) chunks per lane
    union { v16bf v; int4 q[2]; } A;
    A.q[0] = *reinterpret_cast<const int4*>(xrow + k0 + khalf * 16);
    A.q[1] = *reinterpret_cast<const int4*>(xrow + k0 + khalf * 16 + 8);
    // B: 32 contiguous bytes per lane from LDS (pre-packed register image)
    union { v16bf v; int4 q[2]; } B;
    const unsigned short* bp = Bs + (((tn << 2) + kc) * 32 + lane) * 16;
    B.q[0] = *reinterpret_cast<const int4*>(bp);
    B.q[1] = *reinterpret_cast<const int4*>(bp + 8);

    c = __builtin_amdgcn_wmma_f32_16x16x32_bf16(
        false, A.v, false, B.v, (short)0, c, false, false);
  }

  // store D as bf16 (halves gather traffic in the edge phase)
  const int mbase = tm * 16 + khalf * 8;
  #pragma unroll
  for (int j = 0; j < 8; ++j)
    hb[(size_t)(mbase + j) * DH + col] = f2bf(c[j]);
}

// ---------------- layer-1 aggregate ----------------
// a[i,f] = h[i,f]*dinv[i]^2 + b1[f]  (self-loop + bias)
__global__ void k_init_a(const unsigned short* __restrict__ hb, const float* __restrict__ dinv,
                         const float* __restrict__ b1, float* __restrict__ a, int n) {
  int i = blockIdx.x * blockDim.x + threadIdx.x;      // over n*DH
  if (i < n * DH) {
    int node = i >> 6, f = i & 63;
    float di = dinv[node];
    a[i] = bf2f(hb[i]) * di * di + b1[f];
  }
}
// one wave per edge: gather bf16 pair per lane, two f32 atomics
__global__ void k_edge_agg1(const int* __restrict__ src, const int* __restrict__ dst,
                            const float* __restrict__ w, const float* __restrict__ dinv,
                            const unsigned short* __restrict__ hb, float* __restrict__ a, int e) {
  const int wv = (blockIdx.x * blockDim.x + threadIdx.x) >> 5;
  const int lane = threadIdx.x & 31;
  if (wv >= e) return;                                // wave-uniform
  const int s = src[wv];                              // uniform -> scalar loads
  const int d = dst[wv];
  const float nrm = dinv[s] * w[wv] * dinv[d];
  const unsigned int hp =
      reinterpret_cast<const unsigned int*>(hb + (size_t)s * DH)[lane];
  float* ap = a + (size_t)d * DH + lane * 2;
  atomicAdd(ap + 0, bf2f((unsigned short)(hp & 0xffffu)) * nrm);
  atomicAdd(ap + 1, bf2f((unsigned short)(hp >> 16)) * nrm);
}

// ---------------- relu + dropout + dense(64->1) ----------------
__global__ void k_relu_drop_dot(const float* __restrict__ a, const float* __restrict__ W2,
                                float* __restrict__ z, int n) {
  const int wv = (blockIdx.x * blockDim.x + threadIdx.x) >> 5;
  const int lane = threadIdx.x & 31;
  if (wv >= n) return;
  float acc = 0.0f;
  #pragma unroll
  for (int t = 0; t < 2; ++t) {
    const int f = lane * 2 + t;
    float v = a[(size_t)wv * DH + f];
    v = v > 0.0f ? v : 0.0f;                          // relu
    unsigned int hsh = wang_hash((unsigned int)(wv * DH + f) ^ 0x2a2a2a2au);
    v = (hsh & 1u) ? v * 2.0f : 0.0f;                 // dropout p=0.5, scale 2
    acc += v * W2[f];
  }
  #pragma unroll
  for (int off = 16; off > 0; off >>= 1)
    acc += __shfl_xor(acc, off);
  if (lane == 0) z[wv] = acc;
}

// ---------------- layer-2 aggregate ----------------
__global__ void k_init_out(const float* __restrict__ z, const float* __restrict__ dinv,
                           const float* __restrict__ b2, float* __restrict__ out, int n) {
  int i = blockIdx.x * blockDim.x + threadIdx.x;
  if (i < n) {
    float di = dinv[i];
    out[i] = z[i] * di * di + b2[0];
  }
}
__global__ void k_edge_agg2(const int* __restrict__ src, const int* __restrict__ dst,
                            const float* __restrict__ w, const float* __restrict__ dinv,
                            const float* __restrict__ z, float* __restrict__ out, int e) {
  int i = blockIdx.x * blockDim.x + threadIdx.x;
  if (i < e) {
    int s = src[i], d = dst[i];
    atomicAdd(&out[d], z[s] * dinv[s] * w[i] * dinv[d]);
  }
}

extern "C" void kernel_launch(void* const* d_in, const int* in_sizes, int n_in,
                              void* d_out, int out_size, void* d_ws, size_t ws_size,
                              hipStream_t stream) {
  const float* x  = (const float*)d_in[0];
  const int*   ei = (const int*)d_in[1];
  const float* w  = (const float*)d_in[2];
  const float* W1 = (const float*)d_in[3];
  const float* b1 = (const float*)d_in[4];
  const float* W2 = (const float*)d_in[5];
  const float* b2 = (const float*)d_in[6];
  float* out = (float*)d_out;

  const int N = in_sizes[0] / DIN;     // 100000
  const int E = in_sizes[2];           // 1600000
  const int* src = ei;
  const int* dst = ei + E;

  // ---- workspace layout (~39.2 MB) ----
  char* p = (char*)d_ws;
  float* deg  = (float*)p;                 p += (size_t)N * 4;            // also z
  float* dinv = (float*)p;                 p += (size_t)N * 4;
  unsigned short* hb = (unsigned short*)p; p += (size_t)N * DH * 2;       // h in bf16
  // xb (N*DIN*2 bytes) and a (N*DH*4 bytes) are the same size; xb dies before a is written
  unsigned short* xb = (unsigned short*)p;
  float*          a  = (float*)p;          p += (size_t)N * DH * 4;
  unsigned short* Bpack = (unsigned short*)p;                              // 16 KB
  float* z = deg;                          // deg dead after k_dinv

  const int T = 256;
  // normalization
  k_init_deg   <<<(N + T - 1) / T, T, 0, stream>>>(deg, N);
  k_deg_scatter<<<(E + T - 1) / T, T, 0, stream>>>(dst, w, deg, E);
  k_dinv       <<<(N + T - 1) / T, T, 0, stream>>>(deg, dinv, N);

  // prep: bf16 x, packed B image
  k_pack_x<<<((size_t)N * DIN + T - 1) / T, T, 0, stream>>>(x, xb, N * DIN);
  k_pack_B<<<(BPACK_ELEMS + T - 1) / T, T, 0, stream>>>(W1, Bpack);

  // h = x @ W1 via WMMA (one wave per 16x16 tile; 25000 waves)
  const int waves = (N / 16) * (DH / 16);
  k_gemm_wmma<<<(waves * 32 + T - 1) / T, T, 0, stream>>>(xb, Bpack, hb, N);

  // layer-1 aggregate + bias + self-loop
  k_init_a   <<<((size_t)N * DH + T - 1) / T, T, 0, stream>>>(hb, dinv, b1, a, N);
  k_edge_agg1<<<((size_t)E * 32 + T - 1) / T, T, 0, stream>>>(src, dst, w, dinv, hb, a, E);

  // relu + dropout + dense(64->1)
  k_relu_drop_dot<<<((size_t)N * 32 + T - 1) / T, T, 0, stream>>>(a, W2, z, N);

  // layer-2 aggregate
  k_init_out <<<(N + T - 1) / T, T, 0, stream>>>(z, dinv, b2, out, N);
  k_edge_agg2<<<(E + T - 1) / T, T, 0, stream>>>(src, dst, w, dinv, z, out, E);
}
